// Conv2d_Mimic_62019327754449
// MI455X (gfx1250) — matrix operand order repaired and checked by reading, not compiled, not run on hardware
//
#include <hip/hip_runtime.h>
#include <stdint.h>

typedef __bf16 bf16;
typedef __attribute__((ext_vector_type(8)))  __bf16 v8bf;
typedef __attribute__((ext_vector_type(16))) __bf16 v16bf;
typedef __attribute__((ext_vector_type(8)))  float  v8f;

#define C_IN   128
#define C_OUT  256
#define HW     56
#define SP     (HW * HW)        // 3136 pixels per image
#define KTOT   (C_IN * 9)       // 1152
#define NPIX   (32 * SP)        // 100352 output pixels
#define BM     128              // couts per block
#define BP     128              // pixels per block
#define LDSS   40               // LDS row stride in bf16 (32 data + 8 pad = 80B, conflict-free)
#define NKS    36               // K steps: 9 positions x 4 chunks of 32

union FragU { v16bf v; v8bf h[2]; };

// ---------------- weight prepack: fp32 [cout][c*9+p] -> bf16 hi/lo [cout][p*128+c] -------
__global__ void prepack_w(const float* __restrict__ W,
                          bf16* __restrict__ Whi, bf16* __restrict__ Wlo) {
  int d = blockIdx.x * 256 + threadIdx.x;
  if (d >= C_OUT * KTOT) return;
  int cout = d / KTOT;
  int kk   = d - cout * KTOT;
  int p    = kk >> 7;            // kernel position 0..8  (kh*3+kw)
  int c    = kk & 127;           // input channel
  float v  = W[cout * KTOT + c * 9 + p];
  bf16 h   = (bf16)v;                       // RNE high part
  bf16 l   = (bf16)(v - (float)h);          // residual low part
  Whi[d] = h;
  Wlo[d] = l;
}

// A-operand (16x32, 16-bit): lane<16 -> K {off..off+7, off+16..off+23}, off = lhalf*8
static __device__ inline v16bf ld_a(const bf16* rowp, int lhalf) {
  FragU f;
  f.h[0] = *(const v8bf*)(rowp + lhalf * 8);
  f.h[1] = *(const v8bf*)(rowp + lhalf * 8 + 16);
  return f.v;
}
// B-operand (32x16, 16-bit): lane<16 -> K {off..off+7, off+16..off+23}, off = lhalf*8
static __device__ inline v16bf ld_b(const bf16* rowp, int lhalf) {
  FragU f;
  f.h[0] = *(const v8bf*)(rowp + lhalf * 8);
  f.h[1] = *(const v8bf*)(rowp + lhalf * 8 + 16);
  return f.v;
}

__global__ __launch_bounds__(256)
void conv_wmma(const float* __restrict__ x, const bf16* __restrict__ Whi,
               const bf16* __restrict__ Wlo, const float* __restrict__ bias,
               float* __restrict__ out) {
  // double-buffered tiles: 2 * (2 weight + 2 patch) * 10240B = 80 KB
  __shared__ __attribute__((aligned(16))) bf16 sWh[2][BM * LDSS];
  __shared__ __attribute__((aligned(16))) bf16 sWl[2][BM * LDSS];
  __shared__ __attribute__((aligned(16))) bf16 sPh[2][BP * LDSS];
  __shared__ __attribute__((aligned(16))) bf16 sPl[2][BP * LDSS];

  const int tid      = threadIdx.x;
  const int mBase    = blockIdx.x * BP;   // first pixel of block
  const int coutBase = blockIdx.y * BM;   // first cout of block

  // wave tiling: 4 waves along cout (32 each), 2 waves along pixels (64 each)
  const int wave    = tid >> 5;
  const int lane    = tid & 31;
  const int lrow    = lane & 15;
  const int lhalf   = lane >> 4;
  const int coutOff = (wave >> 1) * 32;
  const int pixOff  = (wave & 1) * 64;

  // ---- patch staging geometry: each thread owns one pixel row, 16 contiguous channels ----
  const int prow  = tid & 127;
  const int ccSel = tid >> 7;                  // 0/1: lower/upper 16 channels of the 32-chunk
  const int m     = mBase + prow;
  const int nImg  = m / SP;
  const int po    = m - nImg * SP;
  const int oh    = po / HW;
  const int ow    = po - oh * HW;
  const float* xImg = x + (size_t)nImg * C_IN * SP;

  // ---- weight staging geometry: 4 threads per cout row, 8 bf16 each, 2 row-passes ----
  const int wrow0 = tid >> 2;                  // 0..63
  const int wq    = (tid & 3) * 8;             // 0,8,16,24

  float pv[16];

  // issue next weight tile: async global->LDS (ASYNCcnt path), GVS addressing
  auto weights_issue = [&](int koff, int b) {
    const bf16* gh = Whi + (size_t)coutBase * KTOT + koff;
    const bf16* gl = Wlo + (size_t)coutBase * KTOT + koff;
#pragma unroll
    for (int j = 0; j < 2; ++j) {
      int row = wrow0 + j * 64;
      uint32_t voff = (uint32_t)(row * KTOT + wq) * 2u;               // byte offset
      uint32_t dsh  = (uint32_t)(uintptr_t)&sWh[b][row * LDSS + wq];  // LDS byte offset
      uint32_t dsl  = (uint32_t)(uintptr_t)&sWl[b][row * LDSS + wq];
      asm volatile("global_load_async_to_lds_b128 %0, %1, %2"
                   :: "v"(dsh), "v"(voff), "s"(gh) : "memory");
      asm volatile("global_load_async_to_lds_b128 %0, %1, %2"
                   :: "v"(dsl), "v"(voff), "s"(gl) : "memory");
    }
  };

  // issue next patch tile global loads into registers
  auto patch_issue = [&](int ks) {
    const int p  = ks >> 2;
    const int c0 = (ks & 3) * 32;
    const int kh = p / 3, kw = p - kh * 3;
    const int ih = oh + kh - 1, iw = ow + kw - 1;
    const bool valid = ((unsigned)ih < HW) && ((unsigned)iw < HW);
    const int  sp    = ih * HW + iw;
#pragma unroll
    for (int i = 0; i < 16; ++i) {
      int cc = ccSel * 16 + i;
      pv[i] = valid ? xImg[(c0 + cc) * SP + sp] : 0.0f;
    }
  };

  // convert staged registers to split-bf16 and store as 16B chunks
  auto patch_store = [&](int b) {
#pragma unroll
    for (int g = 0; g < 2; ++g) {
      v8bf hv, lv;
#pragma unroll
      for (int i = 0; i < 8; ++i) {
        float v = pv[g * 8 + i];
        bf16  h = (bf16)v;
        hv[i] = h;
        lv[i] = (bf16)(v - (float)h);
      }
      int off = prow * LDSS + ccSel * 16 + g * 8;
      *(v8bf*)&sPh[b][off] = hv;
      *(v8bf*)&sPl[b][off] = lv;
    }
  };

  v8f acc[2][4];
#pragma unroll
  for (int i = 0; i < 2; ++i)
#pragma unroll
    for (int j = 0; j < 4; ++j)
#pragma unroll
      for (int r = 0; r < 8; ++r) acc[i][j][r] = 0.0f;

  // ---- prologue: stage K-step 0 into buffer 0 ----
  patch_issue(0);
  weights_issue(0, 0);
  patch_store(0);
  asm volatile("s_wait_asynccnt 0" ::: "memory");
  __syncthreads();

  // ---- pipelined K loop: prefetch ks+1 while computing ks ----
  for (int ks = 0; ks < NKS; ++ks) {
    const int cur = ks & 1, nxt = cur ^ 1;
    const bool pre = (ks + 1 < NKS);
    if (pre) {
      const int ks1 = ks + 1;
      patch_issue(ks1);                                    // global loads in flight
      weights_issue((ks1 >> 2) * 128 + (ks1 & 3) * 32, nxt); // async copies in flight
    }

    // 2x4 fragments, split-bf16: hi*hi + hi*lo + lo*hi (fp32 accumulate)
#pragma unroll
    for (int fm = 0; fm < 2; ++fm) {
      const int wr = (coutOff + fm * 16 + lrow) * LDSS;
      v16bf ah = ld_a(&sWh[cur][wr], lhalf);
      v16bf al = ld_a(&sWl[cur][wr], lhalf);
#pragma unroll
      for (int fn = 0; fn < 4; ++fn) {
        const int pr = (pixOff + fn * 16 + lrow) * LDSS;
        v16bf bh = ld_b(&sPh[cur][pr], lhalf);
        v16bf bl = ld_b(&sPl[cur][pr], lhalf);
        acc[fm][fn] = __builtin_amdgcn_wmma_f32_16x16x32_bf16(
            false, ah, false, bh, (short)0, acc[fm][fn], false, false);
        acc[fm][fn] = __builtin_amdgcn_wmma_f32_16x16x32_bf16(
            false, ah, false, bl, (short)0, acc[fm][fn], false, false);
        acc[fm][fn] = __builtin_amdgcn_wmma_f32_16x16x32_bf16(
            false, al, false, bh, (short)0, acc[fm][fn], false, false);
      }
    }

    if (pre) patch_store(nxt);
    asm volatile("s_wait_asynccnt 0" ::: "memory");
    __syncthreads();
  }

  // ---- epilogue: D layout -> lane = pixel (contiguous, coalesced fp32 stores) ----
#pragma unroll
  for (int fn = 0; fn < 4; ++fn) {
    int pix = mBase + pixOff + fn * 16 + lrow;   // 16-pixel frag never crosses image bound
    int ni  = pix / SP;
    int pp  = pix - ni * SP;
#pragma unroll
    for (int fm = 0; fm < 2; ++fm) {
      int cbase = coutBase + coutOff + fm * 16 + lhalf * 8;
#pragma unroll
      for (int r = 0; r < 8; ++r) {
        int cout = cbase + r;
        out[(size_t)(ni * C_OUT + cout) * SP + pp] = acc[fm][fn][r] + bias[cout];
      }
    }
  }
}

extern "C" void kernel_launch(void* const* d_in, const int* in_sizes, int n_in,
                              void* d_out, int out_size, void* d_ws, size_t ws_size,
                              hipStream_t stream) {
  const float* x = (const float*)d_in[0];
  const float* W = (const float*)d_in[1];
  const float* b = (const float*)d_in[2];
  float* out = (float*)d_out;

  bf16* Whi = (bf16*)d_ws;
  bf16* Wlo = Whi + (size_t)C_OUT * KTOT;   // 1.18 MB total in workspace

  prepack_w<<<(C_OUT * KTOT + 255) / 256, 256, 0, stream>>>(W, Whi, Wlo);

  dim3 grid(NPIX / BP, C_OUT / BM);         // 784 x 2 blocks
  conv_wmma<<<grid, 256, 0, stream>>>(x, Whi, Wlo, b, out);
}
